// IGIRNNModel_50027779064234
// MI455X (gfx1250) — matrix operand order repaired and checked by reading, not compiled, not run on hardware
//
#include <hip/hip_runtime.h>

// Problem dims (fixed by reference)
#define IN_DIM   256
#define HID      1024
#define OUT_DIM  256
#define BATCH    128
#define TSTEPS   1024

#define NBLOCKS  32      // persistent scan kernel grid size (2 M-tiles x 16 N-tiles)
#define LDS_PITCH 40     // halves per row: 80B pitch -> conflict-free ds_load_b128

// WMMA fragment types (gfx1250, wave32)
typedef __attribute__((ext_vector_type(16))) __bf16 v16bf;
typedef __attribute__((ext_vector_type(8)))  __bf16 v8bf;
typedef __attribute__((ext_vector_type(8)))  float  v8f;

union AccF  { v8f  v; float f[8]; };
union FragB { v16bf v; v8bf h[2]; };

__device__ __forceinline__ unsigned short f2bf(float f) {
    // round-to-nearest-even f32 -> bf16
    unsigned int u = __float_as_uint(f);
    u += 0x7FFFu + ((u >> 16) & 1u);
    return (unsigned short)(u >> 16);
}

__device__ __forceinline__ float sigmoidf_fast(float x) {
    return 1.0f / (1.0f + __expf(-x));
}

// ---------------------------------------------------------------------------
// Elementwise helpers
// ---------------------------------------------------------------------------
__global__ void cvt_f32_to_bf16_kernel(const float* __restrict__ src,
                                       unsigned short* __restrict__ dst, int n) {
    for (int i = blockIdx.x * blockDim.x + threadIdx.x; i < n;
         i += gridDim.x * blockDim.x)
        dst[i] = f2bf(src[i]);
}

__global__ void zero_f32_kernel(float* __restrict__ dst, int n) {
    for (int i = blockIdx.x * blockDim.x + threadIdx.x; i < n;
         i += gridDim.x * blockDim.x)
        dst[i] = 0.0f;
}

__global__ void init_sync_kernel(unsigned* __restrict__ sync) {
    sync[0] = 0u;   // arrival counter
    sync[1] = 0u;   // generation
}

// ---------------------------------------------------------------------------
// On-device grid barrier (all NBLOCKS blocks resident; sub-us via L2 atomics)
// ---------------------------------------------------------------------------
__device__ __forceinline__ void grid_sync(unsigned* cnt, unsigned* gen) {
    __syncthreads();
    if (threadIdx.x == 0) {
        __threadfence();  // make this block's h_new writes device-visible
        const unsigned g = __atomic_load_n(gen, __ATOMIC_ACQUIRE);
        const unsigned arrived = __atomic_add_fetch(cnt, 1u, __ATOMIC_ACQ_REL);
        if (arrived == (unsigned)NBLOCKS) {
            __atomic_store_n(cnt, 0u, __ATOMIC_RELAXED);
            __atomic_add_fetch(gen, 1u, __ATOMIC_RELEASE);
        } else {
            while (__atomic_load_n(gen, __ATOMIC_ACQUIRE) == g)
                __builtin_amdgcn_s_sleep(2);
        }
        __threadfence();
    }
    __syncthreads();
}

// ---------------------------------------------------------------------------
// Tile staging: 256 threads each move exactly one 16B chunk (64x32 tile)
// ---------------------------------------------------------------------------
__device__ __forceinline__ void stage_f32_tile(unsigned short* __restrict__ dst,
                                               const float* __restrict__ srcBase,
                                               size_t rowStride, int k0, int tid) {
    const int e = tid * 8;              // 2048 halves / 256 threads
    const int r = e >> 5, c = e & 31;
    const float* p = srcBase + (size_t)r * rowStride + k0 + c;
    const float4 v0 = *(const float4*)(p);
    const float4 v1 = *(const float4*)(p + 4);
    ushort4 s0, s1;
    s0.x = f2bf(v0.x); s0.y = f2bf(v0.y); s0.z = f2bf(v0.z); s0.w = f2bf(v0.w);
    s1.x = f2bf(v1.x); s1.y = f2bf(v1.y); s1.z = f2bf(v1.z); s1.w = f2bf(v1.w);
    *(ushort4*)&dst[r * LDS_PITCH + c]     = s0;
    *(ushort4*)&dst[r * LDS_PITCH + c + 4] = s1;
}

__device__ __forceinline__ void stage_bf16_tile(unsigned short* __restrict__ dst,
                                                const unsigned short* __restrict__ srcBase,
                                                size_t rowStride, int k0, int tid) {
    const int e = tid * 8;
    const int r = e >> 5, c = e & 31;
    *(uint4*)&dst[r * LDS_PITCH + c] =
        *(const uint4*)(srcBase + (size_t)r * rowStride + k0 + c);
}

// ---------------------------------------------------------------------------
// Persistent scan kernel: all T steps of
//   gate_h = h @ Wh^T + bh               (K=1024, bf16 WMMA, double-buffered LDS)
//   i_r    = x_t @ Wx[0:H]^T   + bx[:H]  (K=256)
//   i_i    = x_t @ Wx[H:2H]^T  + bx[H:]
//   h'     = ng + sig(i_i)*(h - ng),  ng = tanh(sig(i_r)*gate_h)
// Block tile 64(M) x 64(N); 8 waves, wave = 2 Mtiles x 1 Ntile; grid = 32.
// ---------------------------------------------------------------------------
__global__ __launch_bounds__(256) void gru_scan_kernel(
    float* __restrict__ hping, float* __restrict__ hpong,
    const float* __restrict__ x,
    const unsigned short* __restrict__ WhB,   // [HID][HID] bf16
    const unsigned short* __restrict__ WxB,   // [2*HID][IN_DIM] bf16
    const float* __restrict__ bx, const float* __restrict__ bh,
    unsigned* __restrict__ sync)
{
    __shared__ __align__(16) unsigned short ldsA[2][64 * LDS_PITCH];
    __shared__ __align__(16) unsigned short ldsB[2][64 * LDS_PITCH];

    const int tid  = threadIdx.x;
    const int lane = tid & 31;
    const int wave = tid >> 5;
    const int wm = wave & 1;        // wave M half  -> rows 32*wm
    const int wn = wave >> 1;       // wave N slice -> cols 16*wn
    const int bm = blockIdx.x & 1;
    const int bn = blockIdx.x >> 1;
    const int m0 = bm * 64;
    const int n0 = bn * 64;

    // Fragment lane maps, contiguous k order shared by A and B (16-bit A 16x32 / B 32x16, wave32)
    const int aRow  = lane & 15;
    const int aKoff = (lane < 16) ? 0 : 16;   // halves [aKoff,+16)
    const int bRow  = wn * 16 + (lane & 15);
    const int bKoff = (lane < 16) ? 0 : 16;   // halves [bKoff,+16)

    const int   ngc = n0 + wn * 16 + (lane & 15);
    const float bxr = bx[ngc];
    const float bxi = bx[HID + ngc];
    const float bhn = bh[ngc];

    const unsigned short* WxrB = WxB + (size_t)n0 * IN_DIM;
    const unsigned short* WxiB = WxB + (size_t)(HID + n0) * IN_DIM;
    const unsigned short* WhRows = WhB + (size_t)n0 * HID;

    float* hsrc = hping;
    float* hdst = hpong;

    for (int t = 0; t < TSTEPS; ++t) {
        AccF accH[2], accR[2], accI[2];
#pragma unroll
        for (int i = 0; i < 2; ++i)
#pragma unroll
            for (int r = 0; r < 8; ++r) {
                accH[i].f[r] = 0.0f; accR[i].f[r] = 0.0f; accI[i].f[r] = 0.0f;
            }

        // ---- Pass 1: gate_h over K = HID, double-buffered LDS -------------
        stage_f32_tile(ldsA[0], hsrc + (size_t)m0 * HID, HID, 0, tid);
        stage_bf16_tile(ldsB[0], WhRows, HID, 0, tid);
        int p = 0;
        for (int k0 = 0; k0 < HID; k0 += 32) {
            __syncthreads();
            if (k0 + 32 < HID) {   // prefetch next chunk into the other buffer
                stage_f32_tile(ldsA[p ^ 1], hsrc + (size_t)m0 * HID, HID, k0 + 32, tid);
                stage_bf16_tile(ldsB[p ^ 1], WhRows, HID, k0 + 32, tid);
            }
            FragB bfr;
            bfr.h[0] = *(const v8bf*)&ldsB[p][bRow * LDS_PITCH + bKoff];
            bfr.h[1] = *(const v8bf*)&ldsB[p][bRow * LDS_PITCH + bKoff + 8];
#pragma unroll
            for (int mt = 0; mt < 2; ++mt) {
                const int row = wm * 32 + mt * 16 + aRow;
                FragB afr;
                afr.h[0] = *(const v8bf*)&ldsA[p][row * LDS_PITCH + aKoff];
                afr.h[1] = *(const v8bf*)&ldsA[p][row * LDS_PITCH + aKoff + 8];
                accH[mt].v = __builtin_amdgcn_wmma_f32_16x16x32_bf16(
                    false, afr.v, false, bfr.v, (short)0, accH[mt].v, false, false);
            }
            p ^= 1;
        }

        // ---- Pass 2: i_r / i_i over K = IN_DIM ----------------------------
        const float* xt = x + ((size_t)m0 * TSTEPS + t) * IN_DIM;
        if (t + 1 < TSTEPS)   // warm next timestep's activations toward L2
            __builtin_prefetch(x + (((size_t)(m0 + (tid & 63))) * TSTEPS + t + 1) * IN_DIM, 0, 1);
        for (int k0 = 0; k0 < IN_DIM; k0 += 32) {
            __syncthreads();
            stage_f32_tile(ldsA[0], xt, (size_t)TSTEPS * IN_DIM, k0, tid);
            stage_bf16_tile(ldsB[0], WxrB, IN_DIM, k0, tid);
            stage_bf16_tile(ldsB[1], WxiB, IN_DIM, k0, tid);
            __syncthreads();

            FragB br, bi;
            br.h[0] = *(const v8bf*)&ldsB[0][bRow * LDS_PITCH + bKoff];
            br.h[1] = *(const v8bf*)&ldsB[0][bRow * LDS_PITCH + bKoff + 8];
            bi.h[0] = *(const v8bf*)&ldsB[1][bRow * LDS_PITCH + bKoff];
            bi.h[1] = *(const v8bf*)&ldsB[1][bRow * LDS_PITCH + bKoff + 8];
#pragma unroll
            for (int mt = 0; mt < 2; ++mt) {
                const int row = wm * 32 + mt * 16 + aRow;
                FragB afr;
                afr.h[0] = *(const v8bf*)&ldsA[0][row * LDS_PITCH + aKoff];
                afr.h[1] = *(const v8bf*)&ldsA[0][row * LDS_PITCH + aKoff + 8];
                accR[mt].v = __builtin_amdgcn_wmma_f32_16x16x32_bf16(
                    false, afr.v, false, br.v, (short)0, accR[mt].v, false, false);
                accI[mt].v = __builtin_amdgcn_wmma_f32_16x16x32_bf16(
                    false, afr.v, false, bi.v, (short)0, accI[mt].v, false, false);
            }
        }

        // ---- Epilogue: fused gates + state update -------------------------
        // C/D layout: VGPR r, lanes 0-15 -> (M=r, N=lane); lanes 16-31 -> (M=r+8)
#pragma unroll
        for (int mt = 0; mt < 2; ++mt) {
#pragma unroll
            for (int r = 0; r < 8; ++r) {
                const int mg = m0 + wm * 32 + mt * 16 + r + ((lane < 16) ? 0 : 8);
                const float hprev = hsrc[(size_t)mg * HID + ngc];
                const float gh    = accH[mt].f[r] + bhn;
                const float rg    = sigmoidf_fast(accR[mt].f[r] + bxr);
                const float ig    = sigmoidf_fast(accI[mt].f[r] + bxi);
                const float ngate = tanhf(rg * gh);
                hdst[(size_t)mg * HID + ngc] = ngate + ig * (hprev - ngate);
            }
        }

        // all blocks finish step t before anyone starts t+1
        grid_sync(sync, sync + 1);
        float* tmp = hsrc; hsrc = hdst; hdst = tmp;
    }
}

// ---------------------------------------------------------------------------
// Final FC: out[b][o] = h[b] . Wfc[o] + bfc[o]   (128x256, K=1024, f32)
// ---------------------------------------------------------------------------
__global__ __launch_bounds__(256) void fc_kernel(
    const float* __restrict__ h, const float* __restrict__ Wfc,
    const float* __restrict__ bfc, float* __restrict__ out)
{
    const int idx = blockIdx.x * blockDim.x + threadIdx.x;
    if (idx >= BATCH * OUT_DIM) return;
    const int b = idx >> 8;
    const int o = idx & 255;
    const float* hr = h   + (size_t)b * HID;
    const float* wr = Wfc + (size_t)o * HID;
    float acc = 0.0f;
    for (int k = 0; k < HID; k += 4) {
        const float4 hv = *(const float4*)(hr + k);
        const float4 wv = *(const float4*)(wr + k);
        acc += hv.x * wv.x + hv.y * wv.y + hv.z * wv.z + hv.w * wv.w;
    }
    out[idx] = acc + bfc[o];
}

// ---------------------------------------------------------------------------
extern "C" void kernel_launch(void* const* d_in, const int* in_sizes, int n_in,
                              void* d_out, int out_size, void* d_ws, size_t ws_size,
                              hipStream_t stream) {
    (void)in_sizes; (void)n_in; (void)out_size; (void)ws_size;

    const float* x   = (const float*)d_in[0];  // [B, T, IN]
    const float* Wx  = (const float*)d_in[1];  // [2H, IN]
    const float* bx  = (const float*)d_in[2];  // [2H]
    const float* Wh  = (const float*)d_in[3];  // [H, H]
    const float* bh  = (const float*)d_in[4];  // [H]
    const float* Wfc = (const float*)d_in[5];  // [OUT, H]
    const float* bfc = (const float*)d_in[6];  // [OUT]
    float* out = (float*)d_out;                // [B, OUT]

    // Workspace layout (~4 MB):
    unsigned char* ws = (unsigned char*)d_ws;
    unsigned short* WhB = (unsigned short*)ws;                               // 2 MB
    unsigned short* WxB = (unsigned short*)(ws + (size_t)HID * HID * 2);     // 1 MB
    float* hping = (float*)(ws + (size_t)HID * HID * 2 + (size_t)2 * HID * IN_DIM * 2);
    float* hpong = hping + BATCH * HID;                                      // 2 x 512 KB
    unsigned* sync = (unsigned*)(hpong + BATCH * HID);                       // 2 words

    cvt_f32_to_bf16_kernel<<<512, 256, 0, stream>>>(Wh, WhB, HID * HID);
    cvt_f32_to_bf16_kernel<<<512, 256, 0, stream>>>(Wx, WxB, 2 * HID * IN_DIM);
    zero_f32_kernel<<<128, 256, 0, stream>>>(hping, BATCH * HID);
    init_sync_kernel<<<1, 1, 0, stream>>>(sync);

    // Persistent scan over all T steps with on-device grid sync.
    gru_scan_kernel<<<NBLOCKS, 256, 0, stream>>>(hping, hpong, x, WhB, WxB, bx, bh, sync);

    // TSTEPS even -> final state lands back in hping
    fc_kernel<<<(BATCH * OUT_DIM + 255) / 256, 256, 0, stream>>>(hping, Wfc, bfc, out);
}
